// R2P2_Dynamic_46205258170684
// MI455X (gfx1250) — matrix operand order repaired and checked by reading, not compiled, not run on hardware
//
#include <hip/hip_runtime.h>
#include <hip/hip_bf16.h>

// ---------------- problem constants ----------------
constexpr int B_TOTAL = 131072;
constexpr int T_STEPS = 30;
constexpr int HPAD = 160;   // H=150 padded to WMMA tiles
constexpr int NPAD = 480;   // 3*HPAD (gate-padded 3H)
constexpr int KX   = 64;    // T*2=60 padded
constexpr int KDS  = 224;   // HPAD + 64 (static 50 padded)
constexpr int NO   = 64;    // MLP1 out 50 padded
constexpr int OSTR = 52;    // LDS stride for o
constexpr int BT   = 64;    // batch rows per workgroup

typedef __attribute__((ext_vector_type(16))) __bf16 v16bf;
typedef __attribute__((ext_vector_type(8)))  __bf16 v8bf;
typedef __attribute__((ext_vector_type(8)))  float  v8f;

// A-fragment (16x32 bf16): lane l holds row m=l&15, K runs [k0,k0+8) and
// [k0+16,k0+24), k0=(l>>4)*8  -> two 16B LDS loads with a 32B gap.
__device__ __forceinline__ v16bf loadA(const __bf16* p) {
  v8bf lo = *(const v8bf*)(p);
  v8bf hi = *(const v8bf*)(p + 16);
  v16bf r;
#pragma unroll
  for (int i = 0; i < 8; ++i) { r[i] = lo[i]; r[i + 8] = hi[i]; }
  return r;
}
// B-fragment (32x16 bf16): lane l holds column n=l&15, K runs [k0,k0+8) and
// [k0+16,k0+24), k0=(l>>4)*8 -> two 16B loads from [N][K]-major weight storage.
__device__ __forceinline__ v16bf loadB(const __bf16* p) {
  v8bf lo = *(const v8bf*)(p);
  v8bf hi = *(const v8bf*)(p + 16);
  v16bf r;
#pragma unroll
  for (int i = 0; i < 8; ++i) { r[i] = lo[i]; r[i + 8] = hi[i]; }
  return r;
}
#define WMMA(Af, Bf, C) \
  __builtin_amdgcn_wmma_f32_16x16x32_bf16(false, (Af), false, (Bf), (short)0, (C), false, false)

__device__ __forceinline__ float fsigmoid(float x) { return 1.0f / (1.0f + __expf(-x)); }
__device__ __forceinline__ float ftanh(float x) {
  x = fminf(fmaxf(x, -15.0f), 15.0f);
  float e = __expf(2.0f * x);
  return (e - 1.0f) / (e + 1.0f);
}
__device__ __forceinline__ float fsoftplus(float x) {
  return (x > 20.0f) ? x : log1pf(__expf(x));
}

__global__ __launch_bounds__(256, 1)
void r2p2_fused(const float* __restrict__ gStatic,
                const float* __restrict__ gInitV,
                const float* __restrict__ gInitP,
                const float* __restrict__ gZ,
                const float* __restrict__ gWih,
                const float* __restrict__ gWhh,
                const float* __restrict__ gBih,
                const float* __restrict__ gBhh,
                const float* __restrict__ gW1,
                const float* __restrict__ gB1,
                const float* __restrict__ gW2,
                const float* __restrict__ gB2,
                float* __restrict__ outMu,
                float* __restrict__ outSig,
                float* __restrict__ outX)
{
  // ~292 KB LDS: whole model resident per workgroup (CDNA5: 320KB/WGP)
  __shared__ __align__(16) __bf16 sWih[NPAD * KX];     //  61440 B
  __shared__ __align__(16) __bf16 sWhh[NPAD * HPAD];   // 153600 B
  __shared__ __align__(16) __bf16 sW1 [NO * KDS];      //  28672 B
  __shared__ __align__(16) __bf16 sH  [BT * HPAD];     //  20480 B
  __shared__ __align__(16) __bf16 sX  [BT * KX];       //   8192 B
  __shared__ __align__(16) __bf16 sS  [BT * 64];       //   8192 B
  __shared__ __align__(16) float  sO  [BT * OSTR];     //  13312 B
  __shared__ float sBih[NPAD];
  __shared__ float sBhh[NPAD];
  __shared__ float sB1[NO];
  __shared__ float sW2m[6 * OSTR];
  __shared__ float sB2[8];

  const int  tid = threadIdx.x;
  const long gb  = (long)blockIdx.x * BT;

  // ---- stage weights (f32 -> bf16, tile-padded) + init activations ----
  for (int i = tid; i < NPAD * KX; i += 256) {
    int n = i / KX, k = i % KX;
    int gate = n / HPAD, j = n % HPAD;
    float v = (j < 150 && k < 60) ? gWih[(gate * 150 + j) * 60 + k] : 0.0f;
    sWih[i] = (__bf16)v;
  }
  for (int i = tid; i < NPAD * HPAD; i += 256) {
    int n = i / HPAD, k = i % HPAD;
    int gate = n / HPAD, j = n % HPAD;
    float v = (j < 150 && k < 150) ? gWhh[(gate * 150 + j) * 150 + k] : 0.0f;
    sWhh[i] = (__bf16)v;
  }
  for (int i = tid; i < NO * KDS; i += 256) {
    int n = i / KDS, k = i % KDS;
    bool valid = (n < 50) && (k < 150 || (k >= HPAD && k < HPAD + 50));
    int ksrc = (k < 150) ? k : (k - HPAD + 150);
    float v = valid ? gW1[n * 200 + ksrc] : 0.0f;
    sW1[i] = (__bf16)v;
  }
  for (int i = tid; i < NPAD; i += 256) {
    int gate = i / HPAD, j = i % HPAD;
    sBih[i] = (j < 150) ? gBih[gate * 150 + j] : 0.0f;
    sBhh[i] = (j < 150) ? gBhh[gate * 150 + j] : 0.0f;
  }
  for (int i = tid; i < NO; i += 256) sB1[i] = (i < 50) ? gB1[i] : 0.0f;
  for (int i = tid; i < 6 * OSTR; i += 256) {
    int g = i / OSTR, k = i % OSTR;
    sW2m[i] = (k < 50) ? gW2[g * 50 + k] : 0.0f;
  }
  if (tid < 8) sB2[tid] = (tid < 6) ? gB2[tid] : 0.0f;
  for (int i = tid; i < BT * 64; i += 256) {
    int r = i >> 6, c = i & 63;
    float v = (c < 50) ? gStatic[(gb + r) * 50 + c] : 0.0f;
    sS[i] = (__bf16)v;
  }
  for (int i = tid; i < BT * KX;   i += 256) sX[i] = (__bf16)0.0f;
  for (int i = tid; i < BT * HPAD; i += 256) sH[i] = (__bf16)0.0f;

  // per-row integrator state (one thread per batch row)
  float xp0 = 0.f, xp1 = 0.f, dx0 = 0.f, dx1 = 0.f;
  if (tid < BT) {
    xp0 = gInitP[(gb + tid) * 2 + 0];
    xp1 = gInitP[(gb + tid) * 2 + 1];
    dx0 = gInitV[(gb + tid) * 2 + 0];
    dx1 = gInitV[(gb + tid) * 2 + 1];
  }

  const int lane = tid & 31;
  const int wave = tid >> 5;
  const int rm   = wave & 3;            // batch row-tile (16 rows)
  const int jh   = wave >> 2;           // hidden-column half (5 j-tiles each)
  const int nlo  = lane & 15;
  const int rb   = (lane >> 4) * 8;     // D-matrix row base for this lane
  const int aRow = rm * 16 + nlo;       // A-fragment row
  const int aK0  = (lane >> 4) * 8;     // A-fragment K base
  const int bK0  = (lane >> 4) * 8;     // B-fragment K base

  __syncthreads();

  // static part of ds: A-fragments constant over time -> hoist
  v16bf Ast[2];
#pragma unroll
  for (int kc = 0; kc < 2; ++kc)
    Ast[kc] = loadA(&sS[aRow * 64 + kc * 32 + aK0]);

#pragma unroll 1
  for (int t = 0; t < T_STEPS; ++t) {
    __syncthreads();   // sX / sH writes from previous step visible

    // A-fragments for x_flat (K=64) and h (K=160) — shared across all N-tiles
    v16bf Ax[2], Ah[5];
#pragma unroll
    for (int kc = 0; kc < 2; ++kc) Ax[kc] = loadA(&sX[aRow * KX + kc * 32 + aK0]);
#pragma unroll
    for (int kc = 0; kc < 5; ++kc) Ah[kc] = loadA(&sH[aRow * HPAD + kc * 32 + aK0]);

    float hnew[5][8];
#pragma unroll
    for (int jt = 0; jt < 5; ++jt) {
      const int jg = (jh * 5 + jt) * 16 + nlo;   // padded hidden col 0..159
      v8f aR = {}, aZ = {}, aI = {}, aN = {};
      // input contributions (ir, iz, inn)
#pragma unroll
      for (int kc = 0; kc < 2; ++kc) {
        aR = WMMA(Ax[kc], loadB(&sWih[(jg)            * KX + kc * 32 + bK0]), aR);
        aZ = WMMA(Ax[kc], loadB(&sWih[(HPAD + jg)     * KX + kc * 32 + bK0]), aZ);
        aI = WMMA(Ax[kc], loadB(&sWih[(2 * HPAD + jg) * KX + kc * 32 + bK0]), aI);
      }
      // hidden contributions (hr, hz fused into r/z; hn kept separate)
#pragma unroll
      for (int kc = 0; kc < 5; ++kc) {
        aR = WMMA(Ah[kc], loadB(&sWhh[(jg)            * HPAD + kc * 32 + bK0]), aR);
        aZ = WMMA(Ah[kc], loadB(&sWhh[(HPAD + jg)     * HPAD + kc * 32 + bK0]), aZ);
        aN = WMMA(Ah[kc], loadB(&sWhh[(2 * HPAD + jg) * HPAD + kc * 32 + bK0]), aN);
      }
      const float br = sBih[jg] + sBhh[jg];
      const float bz = sBih[HPAD + jg] + sBhh[HPAD + jg];
      const float bi = sBih[2 * HPAD + jg];
      const float bh = sBhh[2 * HPAD + jg];
#pragma unroll
      for (int v = 0; v < 8; ++v) {
        float r  = fsigmoid(aR[v] + br);
        float zg = fsigmoid(aZ[v] + bz);
        float nn = ftanh(aI[v] + bi + r * (aN[v] + bh));
        float hold = (float)sH[(rm * 16 + rb + v) * HPAD + jg];
        hnew[jt][v] = (1.0f - zg) * nn + zg * hold;
      }
    }
    __syncthreads();   // all reads of old h done
#pragma unroll
    for (int jt = 0; jt < 5; ++jt) {
      const int jg = (jh * 5 + jt) * 16 + nlo;
#pragma unroll
      for (int v = 0; v < 8; ++v)
        sH[(rm * 16 + rb + v) * HPAD + jg] = (__bf16)hnew[jt][v];
    }
    __syncthreads();   // new h visible

    // ---- MLP layer 1: ds(224) @ W1^T -> softplus -> sO ----
    v16bf Ah2[5];
#pragma unroll
    for (int kc = 0; kc < 5; ++kc) Ah2[kc] = loadA(&sH[aRow * HPAD + kc * 32 + aK0]);
#pragma unroll
    for (int q = 0; q < 2; ++q) {
      const int nt  = jh * 2 + q;
      const int col = nt * 16 + nlo;
      v8f acc = {};
#pragma unroll
      for (int kc = 0; kc < 5; ++kc)
        acc = WMMA(Ah2[kc], loadB(&sW1[col * KDS + kc * 32 + bK0]), acc);
#pragma unroll
      for (int kc = 0; kc < 2; ++kc)
        acc = WMMA(Ast[kc], loadB(&sW1[col * KDS + (5 + kc) * 32 + bK0]), acc);
      const float b1v = sB1[col];
      if (col < OSTR) {
#pragma unroll
        for (int v = 0; v < 8; ++v)
          sO[(rm * 16 + rb + v) * OSTR + col] = fsoftplus(acc[v] + b1v);
      }
    }
    __syncthreads();   // sO ready

    // ---- MLP layer 2 + expm(2x2 sym) + sample + integrate ----
    if (tid < BT) {
      float y[6];
#pragma unroll
      for (int g = 0; g < 6; ++g) {
        float s = sB2[g];
#pragma unroll 10
        for (int k = 0; k < 50; ++k) s = fmaf(sW2m[g * OSTR + k], sO[tid * OSTR + k], s);
        y[g] = fsoftplus(s);
      }
      // S = sig_hat + sig_hat^T ; closed-form expm of symmetric 2x2
      float a2  = 2.0f * y[2], bb = y[3] + y[4], c2 = 2.0f * y[5];
      float htr = 0.5f * (a2 + c2);
      float hdf = 0.5f * (a2 - c2);
      float disc = sqrtf(hdf * hdf + bb * bb);
      float d   = fmaxf(disc, 1e-12f);
      float ed  = __expf(d);
      float ie  = 1.0f / ed;
      float ch  = 0.5f * (ed + ie);
      float scnh = (0.5f * (ed - ie)) / d;
      float esc = __expf(htr);
      float m00 = esc * (ch + scnh * hdf);
      float m11 = esc * (ch - scnh * hdf);
      float m01 = esc * (scnh * bb);

      const long po = (long)t * B_TOTAL + gb + tid;
      float z0 = gZ[po * 2 + 0], z1 = gZ[po * 2 + 1];
      float mu0 = xp0 + dx0 + y[0];
      float mu1 = xp1 + dx1 + y[1];
      float x0 = fmaf(m00, z0, fmaf(m01, z1, mu0));
      float x1 = fmaf(m01, z0, fmaf(m11, z1, mu1));

      outMu [po * 2 + 0] = mu0;  outMu [po * 2 + 1] = mu1;
      outSig[po * 4 + 0] = m00;  outSig[po * 4 + 1] = m01;
      outSig[po * 4 + 2] = m01;  outSig[po * 4 + 3] = m11;
      outX  [po * 2 + 0] = x0;   outX  [po * 2 + 1] = x1;

      dx0 = x0 - xp0; dx1 = x1 - xp1;
      xp0 = x0; xp1 = x1;
      sX[tid * KX + 2 * t + 0] = (__bf16)x0;
      sX[tid * KX + 2 * t + 1] = (__bf16)x1;
    }
  }
}

extern "C" void kernel_launch(void* const* d_in, const int* in_sizes, int n_in,
                              void* d_out, int out_size, void* d_ws, size_t ws_size,
                              hipStream_t stream) {
  (void)in_sizes; (void)n_in; (void)out_size; (void)d_ws; (void)ws_size;
  float* out    = (float*)d_out;
  float* outMu  = out;
  float* outSig = out + (size_t)T_STEPS * B_TOTAL * 2;
  float* outX   = out + (size_t)T_STEPS * B_TOTAL * 6;
  dim3 grid(B_TOTAL / BT), block(256);
  hipLaunchKernelGGL(r2p2_fused, grid, block, 0, stream,
                     (const float*)d_in[0],  (const float*)d_in[1],
                     (const float*)d_in[2],  (const float*)d_in[3],
                     (const float*)d_in[4],  (const float*)d_in[5],
                     (const float*)d_in[6],  (const float*)d_in[7],
                     (const float*)d_in[8],  (const float*)d_in[9],
                     (const float*)d_in[10], (const float*)d_in[11],
                     outMu, outSig, outX);
}
